// GATLayer_55259049230988
// MI455X (gfx1250) — matrix operand order repaired and checked by reading, not compiled, not run on hardware
//
#include <hip/hip_runtime.h>
#include <hip/hip_bf16.h>
#include <math.h>

#define B_   4
#define N_   2048
#define D_   256
#define H_   4
#define DH_  64
#define KKMAX 16

typedef __attribute__((ext_vector_type(16))) __bf16 bf16x16;
typedef __attribute__((ext_vector_type(8)))  float  floatx8;

// ---------- bf16 helpers (RNE, via bit ops only) ----------
__device__ inline unsigned short bfbits(float f) {
  union { float f; unsigned u; } a; a.f = f;
  unsigned u = a.u;
  unsigned r = u + 0x7FFFu + ((u >> 16) & 1u);
  return (unsigned short)(r >> 16);
}

// ---------- WMMA wrapper: D = A(16x32 bf16) * B(32x16 bf16) + C(f32) ----------
__device__ inline floatx8 wmma_bf16(bf16x16 a, bf16x16 b, floatx8 c) {
  return __builtin_amdgcn_wmma_f32_16x16x32_bf16(
      /*neg_a=*/false, a, /*neg_b=*/false, b,
      /*c_mod=*/(short)0, c, /*reuse_a=*/false, /*reuse_b=*/false);
}

// A operand (16x32): lane<16 -> row M=lane, K = kbase+{0..7, 16..23}; lane>=16: +8
// Works for global or LDS-backed ushort arrays (two 16B loads).
__device__ inline bf16x16 load_a_bf16(const unsigned short* base, int stride,
                                      int row, int kbase, int lane) {
  int half = lane >> 4;
  const unsigned short* p = base + (size_t)row * stride + kbase + half * 8;
  union { bf16x16 v; uint4 q[2]; } u;
  u.q[0] = *(const uint4*)(p);
  u.q[1] = *(const uint4*)(p + 16);
  return u.v;
}

// B operand (32x16): lane&15 -> col N; per-lane K = kbase+{0..7, 16..23}; lane>=16: +8
__device__ inline bf16x16 load_b_bf16(const unsigned short* base, int stride,
                                      int row, int kbase, int lane) {
  int half = lane >> 4;
  const unsigned short* p = base + (size_t)row * stride + kbase + half * 8;
  union { bf16x16 v; uint4 q[2]; } u;
  u.q[0] = *(const uint4*)(p);
  u.q[1] = *(const uint4*)(p + 16);
  return u.v;
}

__device__ inline floatx8 zero8() {
  floatx8 z = {0.f, 0.f, 0.f, 0.f, 0.f, 0.f, 0.f, 0.f};
  return z;
}

// ============ K0: transpose + bf16-convert the 4 weight matrices ============
__global__ void k_prep_weights(const float* Wq, const float* Wk, const float* Wv,
                               const float* Wo, unsigned short* WT) {
  int e = blockIdx.x * 256 + threadIdx.x;     // 4 * 256 * 256
  int mat = e >> 16, r = (e >> 8) & 255, c = e & 255;
  const float* W = (mat == 0) ? Wq : (mat == 1) ? Wk : (mat == 2) ? Wv : Wo;
  WT[((size_t)mat << 16) + (r << 8) + c] = bfbits(W[c * 256 + r]);  // WT[j][k]=W[k][j]
}

// ============ K1: row norms -> xbf (bf16 x) and xnbf (bf16 x/||x||) ============
__global__ void k_rownorm(const float* x, unsigned short* xbf, unsigned short* xnbf) {
  int row  = blockIdx.x * 8 + (threadIdx.x >> 5);  // over B*N rows, one wave32/row
  int lane = threadIdx.x & 31;
  const float* xr = x + (size_t)row * D_;
  float v[8], ss = 0.f;
#pragma unroll
  for (int i = 0; i < 8; i++) { v[i] = xr[lane * 8 + i]; ss += v[i] * v[i]; }
#pragma unroll
  for (int off = 16; off > 0; off >>= 1) ss += __shfl_xor(ss, off, 32);
  float inv = 1.0f / fmaxf(sqrtf(ss), 1e-12f);
  unsigned short* ob = xbf  + (size_t)row * D_ + lane * 8;
  unsigned short* on = xnbf + (size_t)row * D_ + lane * 8;
#pragma unroll
  for (int i = 0; i < 8; i++) { ob[i] = bfbits(v[i]); on[i] = bfbits(v[i] * inv); }
}

// ============ K2: cosine-sim row block (WMMA) + top-(k+1) per row ============
// One block = 16 rows of one batch. sim row-block (16x2048 f32) lives in LDS.
__global__ void k_sim_topk(const unsigned short* xnbf, int* topk, const int* kptr) {
  extern __shared__ char smem[];
  float* sim   = (float*)smem;                              // 16*2048 f32 = 128 KB
  float* candv = (float*)(smem + 16 * N_ * 4);              // 16*16*KKMAX
  int*   candi = (int*)(candv + 16 * 16 * KKMAX);           // 16*16*KKMAX

  int b = blockIdx.x >> 7, rt = blockIdx.x & 127, row0 = rt * 16;
  int tid = threadIdx.x, wave = tid >> 5, lane = tid & 31;
  const unsigned short* X = xnbf + (size_t)b * N_ * D_;

  bf16x16 A[8];
  int arow = row0 + (lane & 15);
#pragma unroll
  for (int kc = 0; kc < 8; kc++) A[kc] = load_a_bf16(X, D_, arow, kc * 32, lane);

  for (int t = wave; t < 128; t += 8) {                     // 128 col tiles / 8 waves
    floatx8 acc = zero8();
    int brow = t * 16 + (lane & 15);
#pragma unroll
    for (int kc = 0; kc < 8; kc++) {
      bf16x16 Bv = load_b_bf16(X, D_, brow, kc * 32, lane);
      acc = wmma_bf16(A[kc], Bv, acc);
    }
    int c = lane & 15, half = lane >> 4;
#pragma unroll
    for (int i = 0; i < 8; i++) sim[(i + 8 * half) * N_ + t * 16 + c] = acc[i];
  }
  __syncthreads();

  int kk = *kptr + 1; if (kk > KKMAX) kk = KKMAX;
  // phase A: 16 threads/row, each scans 128 cols keeping local top-kk
  int r = tid >> 4, seg = tid & 15;
  float bv[KKMAX]; int bi[KKMAX];
#pragma unroll
  for (int j = 0; j < KKMAX; j++) { bv[j] = -1e30f; bi[j] = -1; }
  const float* srow = sim + r * N_ + seg * 128;
  for (int c2 = 0; c2 < 128; c2++) {
    float v = srow[c2];
    if (v > bv[kk - 1]) {
      int p = kk - 1;
      while (p > 0 && v > bv[p - 1]) { bv[p] = bv[p - 1]; bi[p] = bi[p - 1]; p--; }
      bv[p] = v; bi[p] = seg * 128 + c2;
    }
  }
  float* cv = candv + (r * 16 + seg) * KKMAX;
  int*   ci = candi + (r * 16 + seg) * KKMAX;
  for (int j = 0; j < kk; j++) { cv[j] = bv[j]; ci[j] = bi[j]; }
  __syncthreads();
  // phase B: one thread/row merges 16*kk candidates
  if (tid < 16) {
    float* cvr = candv + tid * 16 * KKMAX;
    int*   cir = candi + tid * 16 * KKMAX;
    int*   out = topk + ((size_t)b * N_ + row0 + tid) * KKMAX;
    for (int j = 0; j < kk; j++) {
      float best = -1e30f; int bpos = 0;
      for (int q2 = 0; q2 < 16; q2++)
        for (int s2 = 0; s2 < kk; s2++) {
          float v = cvr[q2 * KKMAX + s2];
          if (v > best) { best = v; bpos = q2 * KKMAX + s2; }
        }
      out[j] = cir[bpos];
      cvr[bpos] = -1e30f;
    }
  }
}

// ============ K3a/K3b: adjacency bitmask (zero, then symmetric scatter) ============
__global__ void k_zero(unsigned int* p, int nwords) {
  int i = blockIdx.x * 256 + threadIdx.x;
  if (i < nwords) p[i] = 0u;
}
__global__ void k_adj(const int* topk, unsigned int* adj, const int* kptr) {
  int t = blockIdx.x * 256 + threadIdx.x;     // b*N + n
  if (t >= B_ * N_) return;
  int b = t >> 11, n = t & (N_ - 1);
  int kk = *kptr + 1; if (kk > KKMAX) kk = KKMAX;
  unsigned int* A = adj + (size_t)b * N_ * 64;
  const int* tk = topk + (size_t)t * KKMAX;
  for (int j = 0; j < kk; j++) {
    int m = tk[j];
    atomicOr(&A[(size_t)n * 64 + (m >> 5)], 1u << (m & 31));
    atomicOr(&A[(size_t)m * 64 + (n >> 5)], 1u << (n & 31));  // symmetrize
  }
}

// ============ K4: Q/K/V projections (WMMA). V stored transposed for attn·V B-operand ===========
__global__ void k_qkv(const unsigned short* xbf, const unsigned short* WT,
                      unsigned short* qbf, unsigned short* kbf, unsigned short* vt) {
  int bi = blockIdx.x;
  int sel = bi % 3, rt = (bi / 3) & 127, b = bi / (3 * 128);
  int row0 = rt * 16;
  int tid = threadIdx.x, wave = tid >> 5, lane = tid & 31;
  const unsigned short* X = xbf + (size_t)b * N_ * D_;
  const unsigned short* W = WT + (size_t)sel * D_ * D_;     // W^T, bf16

  bf16x16 A[8];
  int arow = row0 + (lane & 15);
#pragma unroll
  for (int kc = 0; kc < 8; kc++) A[kc] = load_a_bf16(X, D_, arow, kc * 32, lane);

  for (int t = wave; t < 16; t += 8) {                      // 16 output col tiles
    floatx8 acc = zero8();
    int brow = t * 16 + (lane & 15);
#pragma unroll
    for (int kc = 0; kc < 8; kc++) {
      bf16x16 Bv = load_b_bf16(W, D_, brow, kc * 32, lane);
      acc = wmma_bf16(A[kc], Bv, acc);
    }
    int c = lane & 15, half = lane >> 4;
#pragma unroll
    for (int i = 0; i < 8; i++) {
      int r = i + 8 * half, col = t * 16 + c, grow = row0 + r;
      unsigned short hv = bfbits(acc[i]);
      if (sel == 0)      qbf[((size_t)b * N_ + grow) * D_ + col] = hv;
      else if (sel == 1) kbf[((size_t)b * N_ + grow) * D_ + col] = hv;
      else               vt[((size_t)b * D_ + col) * N_ + grow] = hv;  // V^T
    }
  }
}

// ============ K5: masked attention for one (b,h,16-row block) ============
// LDS: f32 scores (128 KB) + bf16 exp-scores in A-operand row-major (64 KB) + misc.
__global__ void k_attn(const unsigned short* qbf, const unsigned short* kbf,
                       const unsigned short* vt, const unsigned int* adj,
                       unsigned short* attnout /* bf16 [B][N][256] */) {
  extern __shared__ char smem[];
  float*          sc      = (float*)smem;                        // 16*2048 f32
  unsigned short* sbf     = (unsigned short*)(smem + 131072);    // 16*2048 bf16
  unsigned int*   arow    = (unsigned int*)(smem + 196608);      // 16*64 words
  float*          red     = (float*)(arow + 16 * 64);            // 16*16
  float*          rowstat = red + 256;                           // 16
  float*          pbuf    = rowstat + 16;                        // 4*16*16 partial C

  int bh = blockIdx.x >> 7, rt = blockIdx.x & 127;
  int b = bh >> 2, h = bh & 3, row0 = rt * 16;
  int tid = threadIdx.x, wave = tid >> 5, lane = tid & 31;

  const unsigned int* Aadj = adj + (size_t)b * N_ * 64;
  for (int i = tid; i < 16 * 64; i += 256)
    arow[i] = Aadj[(size_t)(row0 + (i >> 6)) * 64 + (i & 63)];

  const unsigned short* Q = qbf + (size_t)b * N_ * D_;
  const unsigned short* K = kbf + (size_t)b * N_ * D_;
  bf16x16 Aq0 = load_a_bf16(Q, D_, row0 + (lane & 15), h * 64,      lane);
  bf16x16 Aq1 = load_a_bf16(Q, D_, row0 + (lane & 15), h * 64 + 32, lane);
  __syncthreads();

  const float scale = 0.125f;                                 // 1/sqrt(64)
  for (int t = wave; t < 128; t += 8) {
    floatx8 acc = zero8();
    int krow = t * 16 + (lane & 15);
    acc = wmma_bf16(Aq0, load_b_bf16(K, D_, krow, h * 64,      lane), acc);
    acc = wmma_bf16(Aq1, load_b_bf16(K, D_, krow, h * 64 + 32, lane), acc);
    int c = lane & 15, half = lane >> 4;
#pragma unroll
    for (int i = 0; i < 8; i++) {
      int r = i + 8 * half, m = t * 16 + c;
      bool on = (arow[r * 64 + (m >> 5)] >> (m & 31)) & 1u;
      sc[r * N_ + m] = on ? acc[i] * scale : -1e9f;
    }
  }
  __syncthreads();

  // exact softmax over the full 2048-wide row block (16 threads per row)
  int r = tid >> 4, seg = tid & 15;
  const float* srow = sc + r * N_ + seg * 128;
  float mx = -1e30f;
  for (int i = 0; i < 128; i++) mx = fmaxf(mx, srow[i]);
  red[r * 16 + seg] = mx; __syncthreads();
  if (tid < 16) { float m2 = -1e30f; for (int i = 0; i < 16; i++) m2 = fmaxf(m2, red[tid * 16 + i]); rowstat[tid] = m2; }
  __syncthreads();
  float rmax = rowstat[r], s = 0.f;
  unsigned short* erow = sbf + r * N_ + seg * 128;
  for (int i = 0; i < 128; i++) {                 // exp -> bf16 A-operand buffer
    float e = __expf(srow[i] - rmax);
    s += e;
    erow[i] = bfbits(e);
  }
  __syncthreads();
  red[r * 16 + seg] = s; __syncthreads();
  if (tid < 16) { float s2 = 0.f; for (int i = 0; i < 16; i++) s2 += red[tid * 16 + i]; rowstat[tid] = s2; }
  __syncthreads();

  // attn @ V: 4 output d-tiles; waves split K=2048 in two halves.
  // A-operand now comes straight from bf16 LDS: two 16B loads per 16x32 chunk.
  int tile = wave & 3, khalf = wave >> 2;
  floatx8 acc = zero8();
  const unsigned short* Vt = vt + (size_t)b * D_ * N_;
  int vrow = h * 64 + tile * 16 + (lane & 15);
  for (int kc = khalf * 32; kc < khalf * 32 + 32; kc++) {
    bf16x16 Av = load_a_bf16(sbf, N_, lane & 15, kc * 32, lane);
    bf16x16 Bv = load_b_bf16(Vt, N_, vrow, kc * 32, lane);
    acc = wmma_bf16(Av, Bv, acc);
  }
  if (khalf == 1) {
    int c = lane & 15, half = lane >> 4;
#pragma unroll
    for (int i = 0; i < 8; i++) pbuf[tile * 256 + (i + 8 * half) * 16 + c] = acc[i];
  }
  __syncthreads();
  if (khalf == 0) {
    int c = lane & 15, half = lane >> 4;
#pragma unroll
    for (int i = 0; i < 8; i++) {
      int rr = i + 8 * half;
      float v = (acc[i] + pbuf[tile * 256 + rr * 16 + c]) / rowstat[rr];  // deferred /sum
      attnout[((size_t)b * N_ + row0 + rr) * D_ + h * 64 + tile * 16 + c] = bfbits(v);
    }
  }
}

// ============ K6: out @ Wo + bo, residual, LayerNorm ============
__global__ void k_out_ln(const unsigned short* attnout /* bf16 */,
                         const unsigned short* WT,
                         const float* x, const float* bo, const float* gamma,
                         const float* beta, float* y) {
  __shared__ float obuf[16 * 256];
  __shared__ float red[512];
  __shared__ float mu_[16], rs_[16];
  int b = blockIdx.x >> 7, rt = blockIdx.x & 127, row0 = rt * 16;
  int tid = threadIdx.x, wave = tid >> 5, lane = tid & 31;
  const unsigned short* Ain = attnout + (size_t)b * N_ * D_;
  const unsigned short* Wo = WT + (size_t)3 * D_ * D_;        // Wo^T bf16

  bf16x16 A[8];
  int arow = row0 + (lane & 15);
#pragma unroll
  for (int kc = 0; kc < 8; kc++) A[kc] = load_a_bf16(Ain, D_, arow, kc * 32, lane);

  for (int t = wave; t < 16; t += 8) {
    floatx8 acc = zero8();
    int brow = t * 16 + (lane & 15);
#pragma unroll
    for (int kc = 0; kc < 8; kc++) {
      bf16x16 Bv = load_b_bf16(Wo, D_, brow, kc * 32, lane);
      acc = wmma_bf16(A[kc], Bv, acc);
    }
    int c = lane & 15, half = lane >> 4;
#pragma unroll
    for (int i = 0; i < 8; i++) obuf[(i + 8 * half) * 256 + t * 16 + c] = acc[i];
  }
  __syncthreads();

  int r = tid >> 4, seg = tid & 15;
  float s = 0.f, ss = 0.f, vals[16];
  const float* xr = x + ((size_t)b * N_ + row0 + r) * D_ + seg * 16;
  float* ob = obuf + r * 256 + seg * 16;
#pragma unroll
  for (int i = 0; i < 16; i++) {
    float v = ob[i] + bo[seg * 16 + i] + xr[i];               // + bias + residual
    vals[i] = v; s += v; ss += v * v;
  }
  red[r * 16 + seg] = s; red[256 + r * 16 + seg] = ss;
  __syncthreads();
  if (tid < 16) {
    float m = 0.f, q = 0.f;
    for (int i = 0; i < 16; i++) { m += red[tid * 16 + i]; q += red[256 + tid * 16 + i]; }
    m /= 256.f; q = q / 256.f - m * m;
    mu_[tid] = m; rs_[tid] = rsqrtf(q + 1e-5f);
  }
  __syncthreads();
  float mu = mu_[r], rs = rs_[r];
  float* yr = y + ((size_t)b * N_ + row0 + r) * D_ + seg * 16;
#pragma unroll
  for (int i = 0; i < 16; i++)
    yr[i] = (vals[i] - mu) * rs * gamma[seg * 16 + i] + beta[seg * 16 + i];
}

// ================= host launcher =================
extern "C" void kernel_launch(void* const* d_in, const int* in_sizes, int n_in,
                              void* d_out, int out_size, void* d_ws, size_t ws_size,
                              hipStream_t stream) {
  const float* x     = (const float*)d_in[0];
  const float* Wq    = (const float*)d_in[1];
  const float* Wk    = (const float*)d_in[2];
  const float* Wv    = (const float*)d_in[3];
  const float* Wo    = (const float*)d_in[4];
  const float* bo    = (const float*)d_in[5];
  const float* gamma = (const float*)d_in[6];
  const float* beta  = (const float*)d_in[7];
  const int*   kptr  = (const int*)d_in[8];

  char* ws = (char*)d_ws;
  // workspace layout (bytes, all 16B-aligned)
  unsigned short* XBF   = (unsigned short*)(ws + 0);           //  4 MB bf16 x
  unsigned short* XNBF  = (unsigned short*)(ws + 4194304);     //  4 MB bf16 x/||x||
  unsigned short* WT    = (unsigned short*)(ws + 8388608);     // 512 KB 4x W^T bf16
  unsigned short* QBF   = (unsigned short*)(ws + 8912896);     //  4 MB
  unsigned short* KBF   = (unsigned short*)(ws + 13107200);    //  4 MB
  unsigned short* VT    = (unsigned short*)(ws + 17301504);    //  4 MB V^T
  int*            TOPK  = (int*)(ws + 21495808);               // 512 KB
  unsigned int*   ADJ   = (unsigned int*)(ws + 22020096);      //   2 MB bitmask
  unsigned short* AOUT  = (unsigned short*)(ws + 24117248);    //   4 MB bf16

  k_prep_weights<<<1024, 256, 0, stream>>>(Wq, Wk, Wv, Wo, WT);
  k_rownorm<<<B_ * N_ / 8, 256, 0, stream>>>(x, XBF, XNBF);

  size_t simLds = (size_t)16 * N_ * 4 + (size_t)16 * 16 * KKMAX * 8;
  k_sim_topk<<<B_ * (N_ / 16), 256, simLds, stream>>>(XNBF, TOPK, kptr);

  k_zero<<<(B_ * N_ * 64 + 255) / 256, 256, 0, stream>>>(ADJ, B_ * N_ * 64);
  k_adj<<<(B_ * N_ + 255) / 256, 256, 0, stream>>>(TOPK, ADJ, kptr);

  k_qkv<<<B_ * 128 * 3, 256, 0, stream>>>(XBF, WT, QBF, KBF, VT);

  size_t attnLds = 131072 + 65536 + 16 * 64 * 4 + 256 * 4 + 16 * 4 + 1024 * 4;
  k_attn<<<B_ * H_ * (N_ / 16), 256, attnLds, stream>>>(QBF, KBF, VT, ADJ, AOUT);

  k_out_ln<<<B_ * (N_ / 16), 256, 0, stream>>>(AOUT, WT, x, bo, gamma, beta,
                                               (float*)d_out);
}
